// VOOMv2_36919538877207
// MI455X (gfx1250) — matrix operand order repaired and checked by reading, not compiled, not run on hardware
//
#include <hip/hip_runtime.h>
#include <hip/hip_bf16.h>

typedef __bf16 bf16;
typedef __attribute__((ext_vector_type(16))) __bf16 v16bf;
typedef __attribute__((ext_vector_type(8)))  __bf16 v8bf;
typedef __attribute__((ext_vector_type(8)))  float  v8f;

__device__ __forceinline__ bf16 f2bf(float x) { return (bf16)x; }

// A-fragment (16x32 bf16): lane = row, element e at k=kk+16*half+e.
// Two contiguous 16B loads.
__device__ __forceinline__ v16bf load_afrag(const bf16* __restrict__ p)
{
    v8bf a0 = *(const v8bf*)p;
    v8bf a1 = *(const v8bf*)(p + 8);
    return __builtin_shufflevector(a0, a1, 0, 1, 2, 3, 4, 5, 6, 7,
                                   8, 9, 10, 11, 12, 13, 14, 15);
}
// B-fragment (32x16 bf16): lane = column, k = kk + 16*half + e -> one 32B load.
__device__ __forceinline__ v16bf load_bfrag(const bf16* __restrict__ p)
{
    return *(const v16bf*)p;
}
__device__ __forceinline__ v16bf zero16()
{
    v16bf z;
#pragma unroll
    for (int e = 0; e < 16; ++e) z[e] = (bf16)0.0f;
    return z;
}

// ---------------------------------------------------------------------------
// bf16 WMMA GEMM over channel-last activations.
//   A  : [Mpad][K] bf16 (zero padded), K % 32 == 0, Mpad % 16 == 0
//   B  : [nb*N][Cb] bf16 channel-last, uses channels [0,K)
//   Out NHWC: [nb*N][Co]  (f32 or bf16), writes all Mpad channels
//   Out NCHW: [b][Mout][N] f32 (final output)
// Each wave: 16(M) x 64(N) via 4 accumulators sharing the A fragment.
// ---------------------------------------------------------------------------
__global__ void gemm_wmma(const bf16* __restrict__ A, const bf16* __restrict__ Bm,
                          const float* __restrict__ bias,
                          float* __restrict__ OutF, bf16* __restrict__ OutH,
                          int Mpad, int K, int Cb, int Co,
                          int N, int nb, int Mout, int relu, int nchw)
{
    const int NS = 4;
    int ntb    = N / (16 * NS);
    int tilesN = nb * ntb;
    int Mt     = Mpad >> 4;
    int t = blockIdx.x * (blockDim.x >> 5) + (threadIdx.x >> 5);
    if (t >= Mt * tilesN) return;
    int mt = t / tilesN;
    int g  = t - mt * tilesN;
    long n0 = (long)g * (16 * NS);

    int lane = threadIdx.x & 31;
    int half = lane >> 4, r = lane & 15;
    int m0 = mt << 4;

    v8f zf = {0.f, 0.f, 0.f, 0.f, 0.f, 0.f, 0.f, 0.f};
    v8f acc[NS];
#pragma unroll
    for (int j = 0; j < NS; ++j) acc[j] = zf;

    const bf16* Ap = A + (long)(m0 + r) * K;
    const bf16* Bp = Bm + (n0 + r) * (long)Cb + (half << 4);

    for (int kk = 0; kk < K; kk += 32) {
        v16bf av = load_afrag(Ap + kk + (half << 4));
#pragma unroll
        for (int j = 0; j < NS; ++j) {
            v16bf bv = load_bfrag(Bp + kk + (long)j * 16 * Cb);
            acc[j] = __builtin_amdgcn_wmma_f32_16x16x32_bf16(
                false, av, false, bv, (short)0, acc[j], false, false);
        }
    }

#pragma unroll
    for (int j = 0; j < NS; ++j) {
        long n = n0 + j * 16 + r;
        if (!nchw) {
            if (OutH) {
                v8bf ov;
#pragma unroll
                for (int v = 0; v < 8; ++v) {
                    int m = m0 + (half << 3) + v;
                    float x = acc[j][v] + ((bias && m < Mout) ? bias[m] : 0.f);
                    if (relu) x = fmaxf(x, 0.f);
                    ov[v] = f2bf(x);
                }
                *(v8bf*)(OutH + n * (long)Co + m0 + (half << 3)) = ov;
            } else {
                v8f ov;
#pragma unroll
                for (int v = 0; v < 8; ++v) {
                    int m = m0 + (half << 3) + v;
                    float x = acc[j][v] + ((bias && m < Mout) ? bias[m] : 0.f);
                    if (relu) x = fmaxf(x, 0.f);
                    ov[v] = x;
                }
                *(v8f*)(OutF + n * (long)Co + m0 + (half << 3)) = ov;
            }
        } else {
            int b = (int)(n / N);
            long nn = n - (long)b * N;
#pragma unroll
            for (int v = 0; v < 8; ++v) {
                int m = m0 + (half << 3) + v;
                if (m < Mout) {
                    float x = acc[j][v] + (bias ? bias[m] : 0.f);
                    if (relu) x = fmaxf(x, 0.f);
                    OutF[((long)b * Mout + m) * N + nn] = x;
                }
            }
        }
    }
}

// ---------------------------------------------------------------------------
// Implicit-GEMM 2D conv3x3 (pad=1, no bias): bf16 NHWC in -> f32 NHWC out.
//   Wt: [9][64][64] tap-major, X: [nb*H*W][64], Out: [nb*H*W][64]
// ---------------------------------------------------------------------------
__global__ void conv3x3_wmma(const bf16* __restrict__ Wt, const bf16* __restrict__ X,
                             float* __restrict__ Out, int M, int nb, int Hh, int Ww)
{
    const int Cp = 64, NS = 2;
    int N      = Hh * Ww;
    int ntb    = N / (16 * NS);
    int tilesN = nb * ntb;
    int Mt     = M >> 4;
    int t = blockIdx.x * (blockDim.x >> 5) + (threadIdx.x >> 5);
    if (t >= Mt * tilesN) return;
    int mt = t / tilesN;
    int g  = t - mt * tilesN;
    int b  = g / ntb;
    int nbase = (g - b * ntb) * (16 * NS);

    int lane = threadIdx.x & 31;
    int half = lane >> 4, r = lane & 15;
    int m0 = mt << 4;

    int py0[NS], px0[NS];
#pragma unroll
    for (int j = 0; j < NS; ++j) {
        int p = nbase + j * 16 + r;
        py0[j] = p / Ww;
        px0[j] = p - py0[j] * Ww;
    }
    v8f zf = {0.f, 0.f, 0.f, 0.f, 0.f, 0.f, 0.f, 0.f};
    v8f acc[NS];
#pragma unroll
    for (int j = 0; j < NS; ++j) acc[j] = zf;
    v16bf zv = zero16();

    for (int tap = 0; tap < 9; ++tap) {
        int dy = tap / 3 - 1, dx = tap % 3 - 1;
        const bf16* bb[NS];
        bool ok[NS];
#pragma unroll
        for (int j = 0; j < NS; ++j) {
            int py = py0[j] + dy, px = px0[j] + dx;
            ok[j] = ((unsigned)py < (unsigned)Hh) && ((unsigned)px < (unsigned)Ww);
            int pyc = min(max(py, 0), Hh - 1);
            int pxc = min(max(px, 0), Ww - 1);
            bb[j] = X + ((long)b * N + (long)pyc * Ww + pxc) * Cp + (half << 4);
        }
        const bf16* wrow = Wt + ((long)tap * M + m0 + r) * Cp;
#pragma unroll
        for (int kk = 0; kk < 64; kk += 32) {
            v16bf av = load_afrag(wrow + kk + (half << 4));
#pragma unroll
            for (int j = 0; j < NS; ++j) {
                v16bf bv = load_bfrag(bb[j] + kk);
                bv = ok[j] ? bv : zv;
                acc[j] = __builtin_amdgcn_wmma_f32_16x16x32_bf16(
                    false, av, false, bv, (short)0, acc[j], false, false);
            }
        }
    }
#pragma unroll
    for (int j = 0; j < NS; ++j) {
        v8f ov;
#pragma unroll
        for (int v = 0; v < 8; ++v) ov[v] = acc[j][v];
        *(v8f*)(Out + ((long)b * N + nbase + j * 16 + r) * Cp + m0 + (half << 3)) = ov;
    }
}

// ---------------------------------------------------------------------------
// Implicit-GEMM 3D conv3x3x3 (pad=1, +bias): bf16 NHWC in/out.
//   Wt: [27][48][64] tap-major (Cin padded 48->64), X: [nb*NV][64] (pad ch zero)
//   Out: [nb*NV][64], channels 0..47 written.
// ---------------------------------------------------------------------------
__global__ void conv3d_wmma(const bf16* __restrict__ Wt, const bf16* __restrict__ X,
                            const float* __restrict__ bias, bf16* __restrict__ Out,
                            int M, int nb)
{
    const int Cp = 64, NS = 2;
    const int N = 128 * 32 * 128;
    int ntb    = N / (16 * NS);
    int tilesN = nb * ntb;
    int Mt     = M >> 4;
    int t = blockIdx.x * (blockDim.x >> 5) + (threadIdx.x >> 5);
    if (t >= Mt * tilesN) return;
    int mt = t / tilesN;
    int g  = t - mt * tilesN;
    int b  = g / ntb;
    int nbase = (g - b * ntb) * (16 * NS);

    int lane = threadIdx.x & 31;
    int half = lane >> 4, r = lane & 15;
    int m0 = mt << 4;

    int ix0[NS], iy0[NS], iz0[NS];
#pragma unroll
    for (int j = 0; j < NS; ++j) {
        int p = nbase + j * 16 + r;
        ix0[j] = p >> 12;
        iy0[j] = (p >> 7) & 31;
        iz0[j] = p & 127;
    }
    v8f zf = {0.f, 0.f, 0.f, 0.f, 0.f, 0.f, 0.f, 0.f};
    v8f acc[NS];
#pragma unroll
    for (int j = 0; j < NS; ++j) acc[j] = zf;
    v16bf zv = zero16();

#pragma unroll 1
    for (int tap = 0; tap < 27; ++tap) {
        int ax = tap / 9 - 1;
        int ay = (tap / 3) % 3 - 1;
        int az = tap % 3 - 1;
        const bf16* bb[NS];
        bool ok[NS];
#pragma unroll
        for (int j = 0; j < NS; ++j) {
            int jx = ix0[j] + ax, jy = iy0[j] + ay, jz = iz0[j] + az;
            ok[j] = ((unsigned)jx < 128u) && ((unsigned)jy < 32u) && ((unsigned)jz < 128u);
            int jxc = min(max(jx, 0), 127);
            int jyc = min(max(jy, 0), 31);
            int jzc = min(max(jz, 0), 127);
            long sp = ((long)jxc << 12) + (jyc << 7) + jzc;
            bb[j] = X + ((long)b * N + sp) * Cp + (half << 4);
        }
        const bf16* wrow = Wt + ((long)tap * M + m0 + r) * Cp;
#pragma unroll
        for (int kk = 0; kk < 64; kk += 32) {
            v16bf av = load_afrag(wrow + kk + (half << 4));
#pragma unroll
            for (int j = 0; j < NS; ++j) {
                v16bf bv = load_bfrag(bb[j] + kk);
                bv = ok[j] ? bv : zv;
                acc[j] = __builtin_amdgcn_wmma_f32_16x16x32_bf16(
                    false, av, false, bv, (short)0, acc[j], false, false);
            }
        }
    }
#pragma unroll
    for (int j = 0; j < NS; ++j) {
        v8bf ov;
#pragma unroll
        for (int v = 0; v < 8; ++v) {
            int m = m0 + (half << 3) + v;
            ov[v] = f2bf(acc[j][v] + bias[m]);
        }
        *(v8bf*)(Out + ((long)b * N + nbase + j * 16 + r) * Cp + m0 + (half << 3)) = ov;
    }
}

// ---------------------------------------------------------------------------
// Weight conversion / packing kernels
// ---------------------------------------------------------------------------
// src [M][K] f32 -> dst [Mpad][Kpad] bf16, zero padded
__global__ void cvt_pad2d(const float* __restrict__ s, bf16* __restrict__ d,
                          int M, int K, int Mpad, int Kpad)
{
    int i = blockIdx.x * blockDim.x + threadIdx.x;
    if (i >= Mpad * Kpad) return;
    int m = i / Kpad, k = i - m * Kpad;
    d[i] = (m < M && k < K) ? f2bf(s[(long)m * K + k]) : (bf16)0.0f;
}
// conv weights: src [M][Cin][taps] f32 -> dst [taps][M][Cinpad] bf16
__global__ void cvt_conv(const float* __restrict__ s, bf16* __restrict__ d,
                         int M, int Cin, int Cinpad, int taps)
{
    int tot = taps * M * Cinpad;
    int i = blockIdx.x * blockDim.x + threadIdx.x;
    if (i >= tot) return;
    int tap = i / (M * Cinpad);
    int rem = i - tap * (M * Cinpad);
    int m = rem / Cinpad, k = rem - m * Cinpad;
    d[i] = (k < Cin) ? f2bf(s[((long)m * Cin + k) * taps + tap]) : (bf16)0.0f;
}
// cproj.w0 [64][259] (rgb,feat order) -> [64][288] with (feat0..255, rgb256..258, 0)
__global__ void cvt_cp0(const float* __restrict__ s, bf16* __restrict__ d)
{
    int i = blockIdx.x * blockDim.x + threadIdx.x;
    if (i >= 64 * 288) return;
    int m = i / 288, k = i - m * 288;
    float v = 0.f;
    if (k < 256)      v = s[m * 259 + 3 + k];
    else if (k < 259) v = s[m * 259 + (k - 256)];
    d[i] = f2bf(v);
}

// ---------------------------------------------------------------------------
// Input packing (channel-last Xc [nb*N2][288]: feat 0..255, rgb 256..258, 0 pad)
// ---------------------------------------------------------------------------
__global__ void pack_feat(const float* __restrict__ feat, bf16* __restrict__ Xc,
                          int nb, int N2)
{
    const int CS = 288 - 256;                 // handles channels 256.. via resize/zero
    long tot = (long)nb * N2 * 256;
    long i = (long)blockIdx.x * blockDim.x + threadIdx.x;
    if (i >= tot) return;
    int c = (int)(i & 255);                   // 256 channels
    long np = i >> 8;
    int b = (int)(np / N2);
    int n = (int)(np - (long)b * N2);
    Xc[np * 288 + c] = f2bf(feat[((long)b * 256 + c) * N2 + n]);
    (void)CS;
}
__global__ void pack_pad(bf16* __restrict__ Xc, int nb, int N2)
{
    long tot = (long)nb * N2 * 29;            // channels 259..287
    long i = (long)blockIdx.x * blockDim.x + threadIdx.x;
    if (i >= tot) return;
    long np = i / 29;
    int c = (int)(i - np * 29) + 259;
    Xc[np * 288 + c] = (bf16)0.0f;
}
__global__ void resize_pack(const float* __restrict__ rgb, bf16* __restrict__ Xc,
                            int nb, int H, int W, int FH, int FW)
{
    int N2 = FH * FW;
    long tot = (long)nb * N2 * 3;
    long i = (long)blockIdx.x * blockDim.x + threadIdx.x;
    if (i >= tot) return;
    int c = (int)(i % 3);
    long np = i / 3;
    int b = (int)(np / N2);
    int p = (int)(np - (long)b * N2);
    int oy = p / FW, ox = p - oy * FW;
    float ys = oy * ((float)(H - 1) / (float)(FH - 1));
    float xs = ox * ((float)(W - 1) / (float)(FW - 1));
    int y0 = (int)floorf(ys); int y1 = min(y0 + 1, H - 1);
    int x0 = (int)floorf(xs); int x1 = min(x0 + 1, W - 1);
    float wy = ys - (float)y0, wx = xs - (float)x0;
    const float* src = rgb + ((long)b * 3 + c) * (long)H * W;
    float v00 = src[(long)y0 * W + x0], v01 = src[(long)y0 * W + x1];
    float v10 = src[(long)y1 * W + x0], v11 = src[(long)y1 * W + x1];
    float row0 = v00 * (1.f - wx) + v01 * wx;
    float row1 = v10 * (1.f - wx) + v11 * wx;
    Xc[np * 288 + 256 + c] = f2bf(row0 * (1.f - wy) + row1 * wy);
}

// ---------------------------------------------------------------------------
// BatchNorm (training-mode) over channel-last f32 [Ntot][64]: two-pass
// ---------------------------------------------------------------------------
__global__ void bn_part(const float* __restrict__ T, float* __restrict__ part, long Ntot)
{
    int c = threadIdx.x & 63, slot = threadIdx.x >> 6;
    long per = (Ntot + gridDim.x - 1) / gridDim.x;
    long s0 = (long)blockIdx.x * per;
    long s1 = min(s0 + per, Ntot);
    float s = 0.f, q = 0.f;
    for (long n = s0 + slot; n < s1; n += 4) {
        float x = T[n * 64 + c];
        s += x; q += x * x;
    }
    __shared__ float sh[2][4][64];
    sh[0][slot][c] = s; sh[1][slot][c] = q;
    __syncthreads();
    if (slot == 0) {
        for (int k = 1; k < 4; ++k) { s += sh[0][k][c]; q += sh[1][k][c]; }
        part[(long)blockIdx.x * 128 + c] = s;
        part[(long)blockIdx.x * 128 + 64 + c] = q;
    }
}
__global__ void bn_finish(const float* __restrict__ part, float* __restrict__ st,
                          int nblk, long Ntot)
{
    int c = threadIdx.x;
    if (c >= 64) return;
    float s = 0.f, q = 0.f;
    for (int i = 0; i < nblk; ++i) { s += part[(long)i * 128 + c]; q += part[(long)i * 128 + 64 + c]; }
    float m = s / (float)Ntot;
    float v = q / (float)Ntot - m * m;
    st[2 * c] = m;
    st[2 * c + 1] = rsqrtf(v + 1e-5f);
}
// y = relu((x-mean)*invstd*g + be [+ skip]) -> bf16, all channel-last [Ntot][64]
__global__ void bn_apply(const float* __restrict__ T, const float* __restrict__ st,
                         const float* __restrict__ g, const float* __restrict__ be,
                         const bf16* __restrict__ skip, bf16* __restrict__ out, long Ntot)
{
    long tot = Ntot * 64;
    long i = (long)blockIdx.x * blockDim.x + threadIdx.x;
    if (i >= tot) return;
    int c = (int)(i & 63);
    float y = (T[i] - st[2 * c]) * st[2 * c + 1] * g[c] + be[c];
    if (skip) y += (float)skip[i];
    out[i] = f2bf(fmaxf(y, 0.f));
}

// in-place softmax over 128 contiguous depth bins: D [nb*N2][128]
__global__ void softmax_depth(float* __restrict__ D, long tot)
{
    long i = (long)blockIdx.x * blockDim.x + threadIdx.x;
    if (i >= tot) return;
    float* p = D + i * 128;
    float mx = -1e30f;
    for (int d = 0; d < 128; ++d) mx = fmaxf(mx, p[d]);
    float s = 0.f;
    for (int d = 0; d < 128; ++d) s += expf(p[d] - mx);
    float inv = 1.f / s;
    for (int d = 0; d < 128; ++d) p[d] = expf(p[d] - mx) * inv;
}

// lift-splat: ctx f32 [nb*N2][64], dprob f32 [nb*N2][128] -> grid bf16 [nb*NV][64]
__global__ void lift_splat(const float* __restrict__ ctx, const float* __restrict__ dprob,
                           const float* __restrict__ Kin, bf16* __restrict__ grid, int nb)
{
    const long NV = 128L * 32 * 128;
    const int FH = 88, FW = 304, N2 = FH * FW;
    long idx = (long)blockIdx.x * blockDim.x + threadIdx.x;
    if (idx >= (long)nb * NV) return;
    int b = (int)(idx / NV);
    long n = idx - (long)b * NV;
    int ix = (int)(n >> 12);
    int iy = (int)((n >> 7) & 31);
    int iz = (int)(n & 127);
    float X = (ix + 0.5f) * 0.2f - 12.8f;
    float Y = (iy + 0.5f) * 0.2f - 1.0f;
    float Z = (iz + 0.5f) * 0.2f;
    float fx = Kin[b * 9 + 0], fy = Kin[b * 9 + 4];
    float cx = Kin[b * 9 + 2], cy = Kin[b * 9 + 5];
    float Zs = fmaxf(Z, 1e-6f);
    float u = fx * X / Zs + cx;
    float v = fy * Y / Zs + cy;
    bool valid = (Z > 1e-3f) && (u >= 0.f) && (u < 1216.f) && (v >= 0.f) && (v < 352.f);
    int ui = (int)fminf(fmaxf(floorf(u * 0.25f), 0.f), (float)(FW - 1));
    int vi = (int)fminf(fmaxf(floorf(v * 0.25f), 0.f), (float)(FH - 1));
    int db = (int)floorf(Z * 5.0f);
    valid = valid && (db >= 0) && (db < 128);
    db = min(max(db, 0), 127);
    long pp = (long)b * N2 + vi * FW + ui;
    float w = valid ? dprob[pp * 128 + db] : 0.0f;
    const float* cb = ctx + pp * 64;
    bf16* gb = grid + idx * 64;
#pragma unroll
    for (int c0 = 0; c0 < 64; c0 += 8) {
        v8bf ov;
#pragma unroll
        for (int j = 0; j < 8; ++j) ov[j] = f2bf(cb[c0 + j] * w);
        *(v8bf*)(gb + c0) = ov;
    }
}

// ---------------------------------------------------------------------------
// GroupNorm (8 groups x 6 ch) over bf16 channel-last [nb*NV][64], ch<48 valid
// ---------------------------------------------------------------------------
__global__ void gn_part(const bf16* __restrict__ R, float* __restrict__ part, int nb)
{
    const long NV = 128L * 32 * 128;
    int b = blockIdx.x >> 6, seg = blockIdx.x & 63;
    int c = threadIdx.x & 63, slot = threadIdx.x >> 6;
    long per = NV / 64;
    long s0 = (long)b * NV + seg * per;
    long s1 = s0 + per;
    float s = 0.f, q = 0.f;
    for (long n = s0 + slot; n < s1; n += 4) {
        float x = (c < 48) ? (float)R[n * 64 + c] : 0.f;
        s += x; q += x * x;
    }
    __shared__ float sh[2][4][64];
    sh[0][slot][c] = s; sh[1][slot][c] = q;
    __syncthreads();
    if (slot == 0) {
        for (int k = 1; k < 4; ++k) { s += sh[0][k][c]; q += sh[1][k][c]; }
        part[(long)blockIdx.x * 128 + c] = s;
        part[(long)blockIdx.x * 128 + 64 + c] = q;
    }
}
__global__ void gn_finish(const float* __restrict__ part, float* __restrict__ st)
{
    const long NV = 128L * 32 * 128;
    int t = threadIdx.x;
    if (t >= 16) return;
    int b = t >> 3, g = t & 7;
    float s = 0.f, q = 0.f;
    for (int seg = 0; seg < 64; ++seg) {
        long blk = (long)(b * 64 + seg) * 128;
        for (int c = g * 6; c < g * 6 + 6; ++c) { s += part[blk + c]; q += part[blk + 64 + c]; }
    }
    float tot = 6.f * (float)NV;
    float m = s / tot;
    float v = q / tot - m * m;
    st[2 * t] = m;
    st[2 * t + 1] = rsqrtf(v + 1e-5f);
}
// G = relu(G + gn(R)) in-place, channels < 48 only (pad stays zero)
__global__ void gn_apply(const bf16* __restrict__ R, bf16* __restrict__ G,
                         const float* __restrict__ st, const float* __restrict__ gam,
                         const float* __restrict__ bet, int nb)
{
    const long NV = 128L * 32 * 128;
    long tot = (long)nb * NV * 64;
    long i = (long)blockIdx.x * blockDim.x + threadIdx.x;
    if (i >= tot) return;
    int c = (int)(i & 63);
    if (c >= 48) return;
    long np = i >> 6;
    int b = (int)(np / NV);
    int sg = b * 8 + c / 6;
    float nrm = ((float)R[i] - st[2 * sg]) * st[2 * sg + 1] * gam[c] + bet[c];
    float y = fmaxf((float)G[i] + nrm, 0.f);
    G[i] = f2bf(y);
}

// ---------------------------------------------------------------------------
// Host orchestration
// ---------------------------------------------------------------------------
extern "C" void kernel_launch(void* const* d_in, const int* in_sizes, int n_in,
                              void* d_out, int out_size, void* d_ws, size_t ws_size,
                              hipStream_t stream)
{
    (void)in_sizes; (void)n_in; (void)out_size; (void)ws_size;
    const int NB = 2, H = 352, W = 1216, FH = 88, FW = 304;
    const int N2 = FH * FW;            // 26752
    const long NV = 128L * 32 * 128;   // 524288
    const long NT2 = (long)NB * N2;    // 53504

    const float* rgb  = (const float*)d_in[0];
    const float* Kin  = (const float*)d_in[1];
    const float* feat = (const float*)d_in[2];
    auto P = [&](int i) { return (const float*)d_in[i]; };
    // flat param order (insertion order):
    // 3:dp.w0 4:dp.b0 5..10:dp.bb1 11..16:dp.bb2 17:dp.w1 18:dp.b1
    // 19:cp.w0 20:cp.b0 21..26:cp.bb1 27..32:cp.bb2 33:cp.w1 34:cp.b1
    // 35:rp.w 36:rp.b 37:rb.w 38:rb.b 39:rb.g 40:rb.be 41:ro.w 42:ro.b

    char* base = (char*)d_ws;
    size_t off = 0;
    auto alloc = [&](size_t bytes) -> void* {
        void* p = base + off;
        off += (bytes + 255) & ~(size_t)255;
        return p;
    };
    bf16*  Xc    = (bf16*)alloc((size_t)NT2 * 288 * 2);
    bf16*  actA  = (bf16*)alloc((size_t)NT2 * 64 * 2);
    bf16*  actB  = (bf16*)alloc((size_t)NT2 * 64 * 2);
    float* tconv = (float*)alloc((size_t)NT2 * 64 * 4);
    float* partb = (float*)alloc(64 * 128 * 4);
    float* bnst  = (float*)alloc(128 * 4);
    float* depth = (float*)alloc((size_t)NT2 * 128 * 4);
    float* ctx   = (float*)alloc((size_t)NT2 * 64 * 4);
    bf16*  grid  = (bf16*)alloc((size_t)NB * NV * 64 * 2);
    bf16*  gbuf  = (bf16*)alloc((size_t)NB * NV * 64 * 2);
    bf16*  rbuf  = (bf16*)alloc((size_t)NB * NV * 64 * 2);
    float* partg = (float*)alloc(128 * 128 * 4);
    float* gnst  = (float*)alloc(32 * 4);
    bf16*  w_dp0 = (bf16*)alloc(64 * 256 * 2);
    bf16*  w_db1a= (bf16*)alloc(9 * 64 * 64 * 2);
    bf16*  w_db1b= (bf16*)alloc(9 * 64 * 64 * 2);
    bf16*  w_db2a= (bf16*)alloc(9 * 64 * 64 * 2);
    bf16*  w_db2b= (bf16*)alloc(9 * 64 * 64 * 2);
    bf16*  w_dp1 = (bf16*)alloc(128 * 64 * 2);
    bf16*  w_cp0 = (bf16*)alloc(64 * 288 * 2);
    bf16*  w_cb1a= (bf16*)alloc(9 * 64 * 64 * 2);
    bf16*  w_cb1b= (bf16*)alloc(9 * 64 * 64 * 2);
    bf16*  w_cb2a= (bf16*)alloc(9 * 64 * 64 * 2);
    bf16*  w_cb2b= (bf16*)alloc(9 * 64 * 64 * 2);
    bf16*  w_cp1 = (bf16*)alloc(64 * 64 * 2);
    bf16*  w_rp  = (bf16*)alloc(64 * 64 * 2);
    bf16*  w_3d  = (bf16*)alloc(27 * 48 * 64 * 2);
    bf16*  w_ro  = (bf16*)alloc(16 * 64 * 2);

    auto pad2d = [&](const float* s, bf16* d, int M, int K, int Mp, int Kp) {
        int n = Mp * Kp;
        cvt_pad2d<<<(n + 255) / 256, 256, 0, stream>>>(s, d, M, K, Mp, Kp);
    };
    auto convw = [&](const float* s, bf16* d, int M, int Cin, int Cp, int taps) {
        int n = taps * M * Cp;
        cvt_conv<<<(n + 255) / 256, 256, 0, stream>>>(s, d, M, Cin, Cp, taps);
    };
    pad2d(P(3),  w_dp0, 64, 256, 64, 256);
    convw(P(5),  w_db1a, 64, 64, 64, 9);  convw(P(8),  w_db1b, 64, 64, 64, 9);
    convw(P(11), w_db2a, 64, 64, 64, 9);  convw(P(14), w_db2b, 64, 64, 64, 9);
    pad2d(P(17), w_dp1, 128, 64, 128, 64);
    cvt_cp0<<<(64 * 288 + 255) / 256, 256, 0, stream>>>(P(19), w_cp0);
    convw(P(21), w_cb1a, 64, 64, 64, 9);  convw(P(24), w_cb1b, 64, 64, 64, 9);
    convw(P(27), w_cb2a, 64, 64, 64, 9);  convw(P(30), w_cb2b, 64, 64, 64, 9);
    pad2d(P(33), w_cp1, 64, 64, 64, 64);
    pad2d(P(35), w_rp, 48, 64, 64, 64);
    convw(P(37), w_3d, 48, 48, 64, 27);
    pad2d(P(41), w_ro, 4, 48, 16, 64);

    {   // inputs -> Xc channel-last [feat 0..255 | rgb 256..258 | zero 259..287]
        long t1 = NT2 * 256;
        pack_feat<<<(int)((t1 + 255) / 256), 256, 0, stream>>>(feat, Xc, NB, N2);
        long t2 = NT2 * 3;
        resize_pack<<<(int)((t2 + 255) / 256), 256, 0, stream>>>(rgb, Xc, NB, H, W, FH, FW);
        long t3 = NT2 * 29;
        pack_pad<<<(int)((t3 + 255) / 256), 256, 0, stream>>>(Xc, NB, N2);
    }

    auto gemm = [&](const bf16* A, const bf16* Bm, const float* bias,
                    float* OutF, bf16* OutH, int Mpad, int K, int Cb, int Co,
                    int N, int Mout, int relu, int nchw) {
        int tiles = (Mpad / 16) * NB * (N / 64);
        gemm_wmma<<<(tiles + 7) / 8, 256, 0, stream>>>(A, Bm, bias, OutF, OutH,
                                                       Mpad, K, Cb, Co, N, NB,
                                                       Mout, relu, nchw);
    };
    int convTiles  = 4 * NB * (N2 / 32);
    int convBlocks = (convTiles + 7) / 8;
    long el2 = NT2 * 64;
    int elBlk = (int)((el2 + 255) / 256);

    auto run_block = [&](const bf16* w1, const float* g1, const float* b1,
                         const bf16* w2, const float* g2, const float* b2,
                         bf16* inout, bf16* tmp) {
        conv3x3_wmma<<<convBlocks, 256, 0, stream>>>(w1, inout, tconv, 64, NB, FH, FW);
        bn_part<<<64, 256, 0, stream>>>(tconv, partb, NT2);
        bn_finish<<<1, 64, 0, stream>>>(partb, bnst, 64, NT2);
        bn_apply<<<elBlk, 256, 0, stream>>>(tconv, bnst, g1, b1, (const bf16*)nullptr, tmp, NT2);
        conv3x3_wmma<<<convBlocks, 256, 0, stream>>>(w2, tmp, tconv, 64, NB, FH, FW);
        bn_part<<<64, 256, 0, stream>>>(tconv, partb, NT2);
        bn_finish<<<1, 64, 0, stream>>>(partb, bnst, 64, NT2);
        bn_apply<<<elBlk, 256, 0, stream>>>(tconv, bnst, g2, b2, inout, inout, NT2);
    };

    // ---- dproj ----
    gemm(w_dp0, Xc, P(4), nullptr, actA, 64, 256, 288, 64, N2, 64, 1, 0);
    run_block(w_db1a, P(6),  P(7),  w_db1b, P(9),  P(10), actA, actB);
    run_block(w_db2a, P(12), P(13), w_db2b, P(15), P(16), actA, actB);
    gemm(w_dp1, actA, P(18), depth, nullptr, 128, 64, 64, 128, N2, 128, 0, 0);
    softmax_depth<<<(int)((NT2 + 255) / 256), 256, 0, stream>>>(depth, NT2);

    // ---- cproj ----
    gemm(w_cp0, Xc, P(20), nullptr, actA, 64, 288, 288, 64, N2, 64, 1, 0);
    run_block(w_cb1a, P(22), P(23), w_cb1b, P(25), P(26), actA, actB);
    run_block(w_cb2a, P(28), P(29), w_cb2b, P(31), P(32), actA, actB);
    gemm(w_cp1, actA, P(34), ctx, nullptr, 64, 64, 64, 64, N2, 64, 0, 0);

    // ---- lift-splat ----
    {
        long tot = (long)NB * NV;
        lift_splat<<<(int)((tot + 255) / 256), 256, 0, stream>>>(ctx, depth, Kin, grid, NB);
    }

    // ---- 3D refine ----
    // rproj: Mpad=64 so pad channels of gbuf are written as zeros (zero weights/bias)
    gemm(w_rp, grid, P(36), nullptr, gbuf, 64, 64, 64, 64, (int)NV, 48, 0, 0);
    {
        int tiles = 3 * NB * (int)(NV / 32);
        conv3d_wmma<<<(tiles + 7) / 8, 256, 0, stream>>>(w_3d, gbuf, P(38), rbuf, 48, NB);
    }
    gn_part<<<128, 256, 0, stream>>>(rbuf, partg, NB);
    gn_finish<<<1, 16, 0, stream>>>(partg, gnst);
    {
        long tot = (long)NB * NV * 64;
        gn_apply<<<(int)((tot + 255) / 256), 256, 0, stream>>>(rbuf, gbuf, gnst,
                                                               P(39), P(40), NB);
    }
    gemm(w_ro, gbuf, P(42), (float*)d_out, nullptr, 16, 64, 64, 64, (int)NV, 4, 0, 1);
}
